// Gate_14293651161746
// MI455X (gfx1250) — matrix operand order repaired and checked by reading, not compiled, not run on hardware
//
#include <hip/hip_runtime.h>
#include <hip/hip_bf16.h>

// ---- problem constants (match reference) ----
#define T_TOKENS 8192
#define DIM      7168
#define NEXP     256
#define BM       32      // tokens per block
#define NGROUPS  8
#define GSIZE    (NEXP / NGROUPS)   // 32
#define TOPKG    4
#define TOPK     8
#define RSCALE   2.5f

typedef __attribute__((ext_vector_type(16))) __bf16 v16bf;
typedef __attribute__((ext_vector_type(8)))  float  v8f;

__device__ __forceinline__ void pack4(v16bf& v, int base, float4 f) {
    v[base + 0] = (__bf16)f.x;
    v[base + 1] = (__bf16)f.y;
    v[base + 2] = (__bf16)f.z;
    v[base + 3] = (__bf16)f.w;
}

__global__ __launch_bounds__(256)
void gate_route_kernel(const float* __restrict__ x,
                       const float* __restrict__ weight,
                       const float* __restrict__ bias,
                       float* __restrict__ out_w,
                       int*   __restrict__ out_i)
{
    __shared__ float s_logits[BM * NEXP];   // 32 KB: logits, later sigmoid scores
    __shared__ float s_bias[NEXP];

    const int tid  = threadIdx.x;
    const int wave = tid >> 5;          // 0..7
    const int lane = tid & 31;
    const int half = lane >> 4;         // 0/1 (wave32 half)
    const int l16  = lane & 15;

    const int tokBase = blockIdx.x * BM;

    s_bias[tid] = bias[tid];            // blockDim == NEXP == 256

    // ---------------- GEMM: logits[32 tokens][256 experts] ----------------
    // wave w: expert columns [32w, 32w+32), both 16-token M tiles.
    v8f acc00 = {}, acc01 = {}, acc10 = {}, acc11 = {};

    // A per-lane row pointers (M = tokBase + mt*16 + l16)
    const float4* xr0 = (const float4*)(x + (size_t)(tokBase +      l16) * DIM);
    const float4* xr1 = (const float4*)(x + (size_t)(tokBase + 16 + l16) * DIM);
    // B per-lane row pointers (expert e = 32*wave + ntl*16 + l16)
    const float4* wr0 = (const float4*)(weight + (size_t)(wave * 32 +      l16) * DIM);
    const float4* wr1 = (const float4*)(weight + (size_t)(wave * 32 + 16 + l16) * DIM);

    for (int k0 = 0; k0 < DIM; k0 += 32) {
        const int kq = k0 >> 2;

        // A fragment layout (16-bit A 16x32): half 0 lanes hold K {0..7,16..23},
        // half 1 lanes hold K {8..15,24..31}; elements 0..7 = first K-octet.
        float4 a0 = xr0[kq + half * 2 + 0];
        float4 a1 = xr0[kq + half * 2 + 1];
        float4 a2 = xr0[kq + 4 + half * 2 + 0];
        float4 a3 = xr0[kq + 4 + half * 2 + 1];
        float4 c0 = xr1[kq + half * 2 + 0];
        float4 c1 = xr1[kq + half * 2 + 1];
        float4 c2 = xr1[kq + 4 + half * 2 + 0];
        float4 c3 = xr1[kq + 4 + half * 2 + 1];

        // B fragment layout (16-bit B 32x16): lane column = l16,
        // same K order as A: half 0 K {0..7,16..23}, half 1 K {8..15,24..31}.
        float4 b0 = wr0[kq + half * 2 + 0];
        float4 b1 = wr0[kq + half * 2 + 1];
        float4 b2 = wr0[kq + 4 + half * 2 + 0];
        float4 b3 = wr0[kq + 4 + half * 2 + 1];
        float4 d0 = wr1[kq + half * 2 + 0];
        float4 d1 = wr1[kq + half * 2 + 1];
        float4 d2 = wr1[kq + 4 + half * 2 + 0];
        float4 d3 = wr1[kq + 4 + half * 2 + 1];

        v16bf A0, A1, B0, B1;
        pack4(A0, 0, a0); pack4(A0, 4, a1); pack4(A0, 8, a2); pack4(A0, 12, a3);
        pack4(A1, 0, c0); pack4(A1, 4, c1); pack4(A1, 8, c2); pack4(A1, 12, c3);
        pack4(B0, 0, b0); pack4(B0, 4, b1); pack4(B0, 8, b2); pack4(B0, 12, b3);
        pack4(B1, 0, d0); pack4(B1, 4, d1); pack4(B1, 8, d2); pack4(B1, 12, d3);

        acc00 = __builtin_amdgcn_wmma_f32_16x16x32_bf16(false, A0, false, B0, (short)0, acc00, false, false);
        acc01 = __builtin_amdgcn_wmma_f32_16x16x32_bf16(false, A0, false, B1, (short)0, acc01, false, false);
        acc10 = __builtin_amdgcn_wmma_f32_16x16x32_bf16(false, A1, false, B0, (short)0, acc10, false, false);
        acc11 = __builtin_amdgcn_wmma_f32_16x16x32_bf16(false, A1, false, B1, (short)0, acc11, false, false);
    }

    // C/D layout: lane column n = l16 (+ tile offset); VGPR j holds row M = j + 8*half.
    {
        const int e0 = wave * 32 + l16;
        #pragma unroll
        for (int j = 0; j < 8; ++j) {
            const int m0 = 8 * half + j;
            s_logits[(m0)      * NEXP + e0]      = acc00[j];
            s_logits[(m0)      * NEXP + e0 + 16] = acc01[j];
            s_logits[(16 + m0) * NEXP + e0]      = acc10[j];
            s_logits[(16 + m0) * NEXP + e0 + 16] = acc11[j];
        }
    }
    __syncthreads();

    // ---------------- per-token routing (32 threads, one token each) ------
    if (tid < BM) {
        float* row = &s_logits[tid * NEXP];

        // pass 1: sigmoid (stored back), biased score, per-group top-2 sum
        float gscore[NGROUPS];
        #pragma unroll
        for (int g = 0; g < NGROUPS; ++g) {
            float m1 = -3.4e38f, m2 = -3.4e38f;
            for (int i = 0; i < GSIZE; ++i) {
                const int e = g * GSIZE + i;
                const float sc = 1.0f / (1.0f + __expf(-row[e]));
                row[e] = sc;                       // keep unbiased sigmoid
                const float s = sc + s_bias[e];
                if (s > m1) { m2 = m1; m1 = s; }
                else if (s > m2) { m2 = s; }
            }
            gscore[g] = m1 + m2;
        }

        // top-4 groups (ties -> lower index, matching lax.top_k)
        bool gsel[NGROUPS];
        #pragma unroll
        for (int g = 0; g < NGROUPS; ++g) {
            int rank = 0;
            #pragma unroll
            for (int h = 0; h < NGROUPS; ++h)
                rank += (gscore[h] > gscore[g]) ||
                        (gscore[h] == gscore[g] && h < g);
            gsel[g] = (rank < TOPKG);
        }

        // pass 2: top-8 experts over selected groups (biased values)
        float tv[TOPK];
        int   tix[TOPK];
        #pragma unroll
        for (int k = 0; k < TOPK; ++k) { tv[k] = -3.4e38f; tix[k] = 0; }
        for (int g = 0; g < NGROUPS; ++g) {
            if (!gsel[g]) continue;
            for (int i = 0; i < GSIZE; ++i) {
                const int e = g * GSIZE + i;
                const float s = row[e] + s_bias[e];
                if (s > tv[TOPK - 1]) {
                    int p = TOPK - 1;
                    while (p > 0 && s > tv[p - 1]) {
                        tv[p] = tv[p - 1]; tix[p] = tix[p - 1]; --p;
                    }
                    tv[p] = s; tix[p] = e;
                }
            }
        }

        // gather unbiased sigmoid scores, normalize, scale
        float wv[TOPK], wsum = 0.0f;
        #pragma unroll
        for (int k = 0; k < TOPK; ++k) { wv[k] = row[tix[k]]; wsum += wv[k]; }
        const float sc = RSCALE / wsum;

        const size_t t = (size_t)tokBase + tid;
        #pragma unroll
        for (int k = 0; k < TOPK; ++k) {
            out_w[t * TOPK + k] = wv[k] * sc;
            out_i[t * TOPK + k] = tix[k];
        }
    }
}

extern "C" void kernel_launch(void* const* d_in, const int* in_sizes, int n_in,
                              void* d_out, int out_size, void* d_ws, size_t ws_size,
                              hipStream_t stream) {
    (void)in_sizes; (void)n_in; (void)out_size; (void)d_ws; (void)ws_size;
    const float* x      = (const float*)d_in[0];
    const float* weight = (const float*)d_in[1];
    const float* bias   = (const float*)d_in[2];

    float* out_w = (float*)d_out;                                  // [T, 8] fp32 weights
    int*   out_i = (int*)((float*)d_out + (size_t)T_TOKENS * TOPK); // [T, 8] int32 indices

    gate_route_kernel<<<T_TOKENS / BM, 256, 0, stream>>>(x, weight, bias, out_w, out_i);
}
